// TransformerBlock_69836168233265
// MI455X (gfx1250) — matrix operand order repaired and checked by reading, not compiled, not run on hardware
//
#include <hip/hip_runtime.h>
#include <stdint.h>

#define T_TOK   4096
#define DMODEL  1024
#define L_SEQ   1024
#define BATCH   4
#define NHEADS  16
#define HDIM    64
#define DC      128
#define DFF2    4096
#define NEXP    8
#define MAXPAIRS (T_TOK*2)
#define RMS_EPS 1.1920928955078125e-7f

// Async LDS staging via CDNA5 GLOBAL_LOAD_ASYNC_TO_LDS (inline asm).
#define USE_ASYNC_STAGING 1

#if defined(__has_builtin)
#if __has_builtin(__builtin_amdgcn_permlane16)
#define HAVE_PERMLANE16 1
#endif
#endif

typedef __attribute__((ext_vector_type(16))) __bf16 v16bf;
typedef __attribute__((ext_vector_type(8)))  float  v8f;
typedef __attribute__((address_space(3))) unsigned short lds_us_t;

union Frag { v16bf v; uint4 q[2]; unsigned short s[16]; };
union Acc  { v8f v; float f[8]; };

__device__ __forceinline__ unsigned short f2bf(float x){
  union { float f; unsigned u; } c; c.f = x;
  unsigned r = c.u + 0x7FFFu + ((c.u >> 16) & 1u);
  return (unsigned short)(r >> 16);
}
__device__ __forceinline__ float bf2f(unsigned short h){
  union { unsigned u; float f; } c; c.u = ((unsigned)h) << 16; return c.f;
}
__device__ __forceinline__ v8f wmma_bf16(v16bf a, v16bf b, v8f c){
  return __builtin_amdgcn_wmma_f32_16x16x32_bf16(false, a, false, b, (short)0, c, false, false);
}
__device__ __forceinline__ unsigned lds_off(const unsigned short* p){
  return (unsigned)(size_t)(lds_us_t*)(unsigned short*)p;
}
__device__ __forceinline__ void async_ld_b128(const unsigned short* lds_dst,
                                              const unsigned short* gsrc){
  unsigned la = lds_off(lds_dst);
  unsigned long long ga = (unsigned long long)gsrc;
  asm volatile("global_load_async_to_lds_b128 %0, %1, off"
               :: "v"(la), "v"(ga) : "memory");
}

// XOR-butterfly within 16-lane halves. v_permlane16_b32 (VALU, no DS latency)
// when available; __shfl_xor (ds_bpermute) fallback. m must be 1/2/4/8.
template<int M>
__device__ __forceinline__ float xor16(float v){
#if HAVE_PERMLANE16
  unsigned sel0, sel1;
  if (M == 1)      { sel0 = 0x67452301u; sel1 = 0xEFCDAB89u; }
  else if (M == 2) { sel0 = 0x54761032u; sel1 = 0xDCFE98BAu; }
  else if (M == 4) { sel0 = 0x32107654u; sel1 = 0xBA98FEDCu; }
  else             { sel0 = 0xFEDCBA98u; sel1 = 0x76543210u; }
  union { float f; unsigned u; } c; c.f = v;
  c.u = __builtin_amdgcn_permlane16(c.u, c.u, sel0, sel1, false, false);
  return c.f;
#else
  return __shfl_xor(v, M, 32);
#endif
}

// ---------------------------------------------------------------------------
// fp32 -> bf16 bulk convert (one-time weight conversion per launch)
// ---------------------------------------------------------------------------
__global__ __launch_bounds__(256)
void f32_to_bf16_kernel(const float* __restrict__ src, unsigned short* __restrict__ dst, int n)
{
  int stride = gridDim.x * 256;
  for (int i = blockIdx.x * 256 + threadIdx.x; i < n; i += stride)
    dst[i] = f2bf(src[i]);
}

// ---------------------------------------------------------------------------
// RMSNorm: out_bf16 = rmsnorm(x)*w ; optionally copy x (fp32) to copyOut.
// ---------------------------------------------------------------------------
__global__ __launch_bounds__(256)
void rmsnorm_kernel(const float* __restrict__ x, const float* __restrict__ w,
                    unsigned short* __restrict__ out, float* __restrict__ copyOut)
{
  int t = blockIdx.x;
  const float* row = x + (size_t)t * DMODEL;
  int lane = threadIdx.x & 31, wave = threadIdx.x >> 5;
  float ss = 0.f;
  for (int i = threadIdx.x; i < DMODEL; i += 256) { float v = row[i]; ss += v*v; }
  #pragma unroll
  for (int d = 1; d < 32; d <<= 1) ss += __shfl_xor(ss, d, 32);
  __shared__ float ps[8];
  if (lane == 0) ps[wave] = ss;
  __syncthreads();
  float tot = 0.f;
  #pragma unroll
  for (int i = 0; i < 8; i++) tot += ps[i];
  float inv = rsqrtf(tot * (1.0f / DMODEL) + RMS_EPS);
  for (int i = threadIdx.x; i < DMODEL; i += 256) {
    float v = row[i];
    out[(size_t)t * DMODEL + i] = f2bf(v * inv * w[i]);
    if (copyOut) copyOut[(size_t)t * DMODEL + i] = v;
  }
}

// ---------------------------------------------------------------------------
// Tiled bf16 WMMA GEMM: C[M,N] = A[M,K](bf16) * W[N,K](bf16)^T
// Block 128 thr (4 waves). Tile 128x64, BK=32; wave -> 32 M-rows, 8 wmma/step.
// Staging via GLOBAL_LOAD_ASYNC_TO_LDS_B128 (ASYNCcnt).
// EPI 0: store bf16            EPI 1: fp32 store + residual add
// EPI 2: MoE up (gathered A rows, SiLU -> bf16 at slot rows)
// EPI 3: MoE down (A rows = slots, scaled atomicAdd into fp32 out)
// ---------------------------------------------------------------------------
template<int EPI>
__global__ __launch_bounds__(128)
void gemm_bf16(const unsigned short* __restrict__ A, const unsigned short* __restrict__ W,
               int M, int N, int K,
               unsigned short* __restrict__ outB, float* __restrict__ outF,
               const float* __restrict__ resid,
               const int* __restrict__ slot_token, const float* __restrict__ slot_w,
               const int* __restrict__ ecounts, const int* __restrict__ eoffs)
{
  int e = blockIdx.z;
  int Mloc = M;
  const unsigned short* Wp = W;
  int obase = 0;
  if (EPI == 2 || EPI == 3) {
    Mloc = ecounts[e];
    Wp = W + (size_t)e * N * K;
    obase = eoffs[e];
  }
  int mbase = blockIdx.y * 128;
  if (mbase >= Mloc) return;
  int nbase = blockIdx.x * 64;

  __shared__ __align__(16) unsigned short As[128][32];
  __shared__ __align__(16) unsigned short Bs[64][32];

  int tid = threadIdx.x, wave = tid >> 5, lane = tid & 31;
  int l16 = lane >> 4, lm = lane & 15;

  Acc acc[2][4];
  #pragma unroll
  for (int s = 0; s < 2; s++)
    #pragma unroll
    for (int j = 0; j < 4; j++)
      #pragma unroll
      for (int r = 0; r < 8; r++) acc[s][j].f[r] = 0.f;

  for (int k0 = 0; k0 < K; k0 += 32) {
    __syncthreads();
    // ---- stage A tile: 128 rows x 32 K, 2 (row,seg) tasks/thread
    #pragma unroll
    for (int i = 0; i < 2; i++) {
      int idx = i * 128 + tid;
      int r = idx >> 1, seg = idx & 1;
      int gr = mbase + r;
      if (gr < Mloc) {
        size_t arow;
        if (EPI == 2)      arow = (size_t)slot_token[obase + gr];
        else if (EPI == 3) arow = (size_t)(obase + gr);
        else               arow = (size_t)gr;
        const unsigned short* src = A + arow * (size_t)K + k0 + seg * 16;
#if USE_ASYNC_STAGING
        async_ld_b128(&As[r][seg * 16], src);
        async_ld_b128(&As[r][seg * 16 + 8], src + 8);
#else
        *(uint4*)&As[r][seg * 16]     = ((const uint4*)src)[0];
        *(uint4*)&As[r][seg * 16 + 8] = ((const uint4*)src)[1];
#endif
      }
    }
    // ---- stage W tile: 64 rows x 32 K, 1 task/thread
    {
      int r = tid >> 1, seg = tid & 1;
      const unsigned short* src = Wp + (size_t)(nbase + r) * K + k0 + seg * 16;
#if USE_ASYNC_STAGING
      async_ld_b128(&Bs[r][seg * 16], src);
      async_ld_b128(&Bs[r][seg * 16 + 8], src + 8);
#else
      *(uint4*)&Bs[r][seg * 16]     = ((const uint4*)src)[0];
      *(uint4*)&Bs[r][seg * 16 + 8] = ((const uint4*)src)[1];
#endif
    }
#if USE_ASYNC_STAGING
    asm volatile("s_wait_asynccnt 0x0" ::: "memory");
#endif
    __syncthreads();
    // ---- compute: wave covers rows [wave*32, wave*32+32)
    Frag a0, a1;
    a0.q[0] = *(const uint4*)&As[wave * 32 + lm][8 * l16];
    a0.q[1] = *(const uint4*)&As[wave * 32 + lm][16 + 8 * l16];
    a1.q[0] = *(const uint4*)&As[wave * 32 + 16 + lm][8 * l16];
    a1.q[1] = *(const uint4*)&As[wave * 32 + 16 + lm][16 + 8 * l16];
    #pragma unroll
    for (int j = 0; j < 4; j++) {
      Frag b;
      b.q[0] = *(const uint4*)&Bs[j * 16 + lm][8 * l16];
      b.q[1] = *(const uint4*)&Bs[j * 16 + lm][16 + 8 * l16];
      acc[0][j].v = wmma_bf16(a0.v, b.v, acc[0][j].v);
      acc[1][j].v = wmma_bf16(a1.v, b.v, acc[1][j].v);
    }
  }

  #pragma unroll
  for (int s = 0; s < 2; s++) {
    #pragma unroll
    for (int j = 0; j < 4; j++) {
      #pragma unroll
      for (int r = 0; r < 8; r++) {
        int mloc = mbase + wave * 32 + s * 16 + r + 8 * l16;
        int col  = nbase + j * 16 + lm;
        if (mloc >= Mloc) continue;
        float v = acc[s][j].f[r];
        if (EPI == 0) {
          outB[(size_t)mloc * N + col] = f2bf(v);
        } else if (EPI == 1) {
          outF[(size_t)mloc * N + col] = v + resid[(size_t)mloc * N + col];
        } else if (EPI == 2) {
          float sv = v / (1.0f + __expf(-v));
          outB[(size_t)(obase + mloc) * N + col] = f2bf(sv);
        } else { // EPI == 3
          int slot = obase + mloc;
          atomicAdd(&outF[(size_t)slot_token[slot] * DMODEL + col], v * slot_w[slot]);
        }
      }
    }
  }
}

// ---------------------------------------------------------------------------
// RoPE fuse: q = rope(q_r)*scale ; ksum = k_c + rope(k_r)
// ---------------------------------------------------------------------------
__global__ __launch_bounds__(256)
void rope_kernel(const unsigned short* __restrict__ qr, const unsigned short* __restrict__ kr,
                 const unsigned short* __restrict__ kc,
                 const float* __restrict__ tsq, const float* __restrict__ tsk,
                 unsigned short* __restrict__ qout, unsigned short* __restrict__ ksum)
{
  int idx = blockIdx.x * 256 + threadIdx.x;          // T_TOK*NHEADS*32
  int i = idx & 31;
  int h = (idx >> 5) & (NHEADS - 1);
  int t = idx >> 9;
  if (t >= T_TOK) return;
  int b = t >> 10, l = t & (L_SEQ - 1);
  float tq = tsq[b * L_SEQ + l];
  float tk = tsk[b * L_SEQ + l];
  float f  = powf(10000.0f, -(float)i * (1.0f / 32.0f));
  float cq = cosf(tq * f), ck = cosf(tk * f);
  size_t base = (size_t)t * DMODEL + h * HDIM + i;
  float q1 = bf2f(qr[base]), q2 = bf2f(qr[base + 32]);
  float k1 = bf2f(kr[base]), k2 = bf2f(kr[base + 32]);
  const float sc = 0.125f;                           // 1/sqrt(64)
  qout[base]      = f2bf((q1 * cq - q2 * ck) * sc);
  qout[base + 32] = f2bf((q2 * cq + q1 * ck) * sc);
  ksum[base]      = f2bf(bf2f(kc[base])      + (k1 * cq - k2 * ck));
  ksum[base + 32] = f2bf(bf2f(kc[base + 32]) + (k2 * cq + k1 * ck));
}

// ---------------------------------------------------------------------------
// V transpose: vt[b][h][d][l] = v_c[(b,l)][h*64+d]
// ---------------------------------------------------------------------------
__global__ void transpose_v_kernel(const unsigned short* __restrict__ vc,
                                   unsigned short* __restrict__ vt)
{
  __shared__ unsigned short tile[16][17];
  int bh = blockIdx.z, b = bh >> 4, h = bh & 15;
  int lbase = blockIdx.x * 16, dbase = blockIdx.y * 16;
  int tx = threadIdx.x, ty = threadIdx.y;
  tile[ty][tx] = vc[((size_t)(b * L_SEQ + lbase + ty)) * DMODEL + h * HDIM + dbase + tx];
  __syncthreads();
  vt[((size_t)(bh * HDIM + dbase + ty)) * L_SEQ + lbase + tx] = tile[tx][ty];
}

// ---------------------------------------------------------------------------
// Flash attention (non-causal). Row-max via v_permlane16 butterfly; row-sum of
// P via an extra WMMA against an all-ones B fragment (C-layout registers).
// ---------------------------------------------------------------------------
__global__ __launch_bounds__(128)
void attn_kernel(const unsigned short* __restrict__ q, const unsigned short* __restrict__ ksum,
                 const unsigned short* __restrict__ vt, unsigned short* __restrict__ ao)
{
  int bh = blockIdx.y, b = bh / NHEADS, h = bh % NHEADS;
  int wave = threadIdx.x >> 5, lane = threadIdx.x & 31;
  int l16 = lane >> 4, lm = lane & 15;
  int qbase = blockIdx.x * 64 + wave * 16;

  __shared__ __align__(16) unsigned short P[4][16][32];

  Frag qa[2];
  const unsigned short* qrow = q + ((size_t)(b * L_SEQ + qbase + lm)) * DMODEL + h * HDIM;
  #pragma unroll
  for (int c = 0; c < 2; c++) {
    qa[c].q[0] = *(const uint4*)(qrow + c * 32 + 8 * l16);
    qa[c].q[1] = *(const uint4*)(qrow + c * 32 + 16 + 8 * l16);
  }
  Frag ones;
  #pragma unroll
  for (int i = 0; i < 16; i++) ones.s[i] = 0x3F80;   // bf16 1.0

  Acc o[4];
  float rmax[8], rsum[8];
  #pragma unroll
  for (int r = 0; r < 8; r++) { rmax[r] = -1e30f; rsum[r] = 0.f; }
  #pragma unroll
  for (int j = 0; j < 4; j++)
    #pragma unroll
    for (int r = 0; r < 8; r++) o[j].f[r] = 0.f;

  const unsigned short* Kb = ksum + (size_t)b * L_SEQ * DMODEL + h * HDIM;
  const unsigned short* Vb = vt + (size_t)bh * HDIM * L_SEQ;

  for (int kb = 0; kb < L_SEQ; kb += 32) {
    Acc S[2];
    #pragma unroll
    for (int ns = 0; ns < 2; ns++) {
      int key = kb + ns * 16 + lm;
      const unsigned short* kr = Kb + (size_t)key * DMODEL;
      Frag b0, b1;
      b0.q[0] = *(const uint4*)(kr + 8 * l16);
      b0.q[1] = *(const uint4*)(kr + 16 + 8 * l16);
      b1.q[0] = *(const uint4*)(kr + 32 + 8 * l16);
      b1.q[1] = *(const uint4*)(kr + 32 + 16 + 8 * l16);
      Acc z;
      #pragma unroll
      for (int r = 0; r < 8; r++) z.f[r] = 0.f;
      S[ns].v = wmma_bf16(qa[0].v, b0.v, z.v);
      S[ns].v = wmma_bf16(qa[1].v, b1.v, S[ns].v);
    }
    // running max (16-lane permlane butterfly), rescale, exponentiate
    float fac[8];
    #pragma unroll
    for (int r = 0; r < 8; r++) {
      float mx = fmaxf(S[0].f[r], S[1].f[r]);
      mx = fmaxf(mx, xor16<1>(mx));
      mx = fmaxf(mx, xor16<2>(mx));
      mx = fmaxf(mx, xor16<4>(mx));
      mx = fmaxf(mx, xor16<8>(mx));
      float nmax = fmaxf(rmax[r], mx);
      float fc = __expf(rmax[r] - nmax);
      S[0].f[r] = __expf(S[0].f[r] - nmax);
      S[1].f[r] = __expf(S[1].f[r] - nmax);
      rmax[r] = nmax; fac[r] = fc;
      rsum[r] *= fc;
    }
    #pragma unroll
    for (int j = 0; j < 4; j++)
      #pragma unroll
      for (int r = 0; r < 8; r++) o[j].f[r] *= fac[r];
    // P: C-layout -> LDS -> A-fragment (in-order DS within wave)
    #pragma unroll
    for (int r = 0; r < 8; r++) {
      P[wave][r + 8 * l16][lm]      = f2bf(S[0].f[r]);
      P[wave][r + 8 * l16][16 + lm] = f2bf(S[1].f[r]);
    }
    asm volatile("s_wait_dscnt 0" ::: "memory");
    Frag pf;
    pf.q[0] = *(const uint4*)&P[wave][lm][8 * l16];
    pf.q[1] = *(const uint4*)&P[wave][lm][16 + 8 * l16];
    // block row-sum of P via WMMA against ones
    Acc z, bs;
    #pragma unroll
    for (int r = 0; r < 8; r++) z.f[r] = 0.f;
    bs.v = wmma_bf16(pf.v, ones.v, z.v);
    #pragma unroll
    for (int r = 0; r < 8; r++) rsum[r] += bs.f[r];
    // PV
    #pragma unroll
    for (int j = 0; j < 4; j++) {
      const unsigned short* vr = Vb + (size_t)(j * 16 + lm) * L_SEQ + kb + 8 * l16;
      Frag vb;
      vb.q[0] = *(const uint4*)vr;
      vb.q[1] = *(const uint4*)(vr + 16);
      o[j].v = wmma_bf16(pf.v, vb.v, o[j].v);
    }
  }
  #pragma unroll
  for (int j = 0; j < 4; j++)
    #pragma unroll
    for (int r = 0; r < 8; r++) {
      int row = qbase + r + 8 * l16;
      int col = h * HDIM + j * 16 + lm;
      ao[((size_t)(b * L_SEQ + row)) * DMODEL + col] = f2bf(o[j].f[r] / rsum[r]);
    }
}

// ---------------------------------------------------------------------------
// Router: one wave per token.
// ---------------------------------------------------------------------------
__global__ __launch_bounds__(32)
void router_kernel(const unsigned short* __restrict__ hn, const float* __restrict__ gw,
                   const float* __restrict__ ebias,
                   int* __restrict__ top_i, float* __restrict__ top_w,
                   int* __restrict__ counts, float* __restrict__ outCounts)
{
  int t = blockIdx.x, lane = threadIdx.x;
  float logit[NEXP];
  #pragma unroll
  for (int e = 0; e < NEXP; e++) {
    float a = 0.f;
    for (int d = lane; d < DMODEL; d += 32)
      a += bf2f(hn[(size_t)t * DMODEL + d]) * gw[(size_t)e * DMODEL + d];
    #pragma unroll
    for (int m = 1; m < 32; m <<= 1) a += __shfl_xor(a, m, 32);
    logit[e] = a + ebias[e];
  }
  if (lane == 0) {
    int i0 = 0;
    #pragma unroll
    for (int e = 1; e < NEXP; e++) if (logit[e] > logit[i0]) i0 = e;
    int i1 = (i0 == 0) ? 1 : 0;
    #pragma unroll
    for (int e = 0; e < NEXP; e++) if (e != i0 && logit[e] > logit[i1]) i1 = e;
    float e1 = __expf(logit[i1] - logit[i0]);
    float inv = 1.0f / (1.0f + e1);
    top_i[t * 2] = i0;  top_i[t * 2 + 1] = i1;
    top_w[t * 2] = inv; top_w[t * 2 + 1] = e1 * inv;
    atomicAdd(&counts[i0], 1); atomicAdd(&counts[i1], 1);
    atomicAdd(&outCounts[i0], 1.0f); atomicAdd(&outCounts[i1], 1.0f);
  }
}

__global__ void scan_kernel(const int* __restrict__ counts, int* __restrict__ offs)
{
  if (threadIdx.x == 0) {
    int a = 0;
    for (int e = 0; e < NEXP; e++) { offs[e] = a; a += counts[e]; }
  }
}

__global__ __launch_bounds__(256)
void scatter_kernel(const int* __restrict__ top_i, const float* __restrict__ top_w,
                    const int* __restrict__ offs, int* __restrict__ fill,
                    int* __restrict__ slot_token, float* __restrict__ slot_w)
{
  int t = blockIdx.x * 256 + threadIdx.x;
  if (t >= T_TOK) return;
  #pragma unroll
  for (int k = 0; k < 2; k++) {
    int e = top_i[t * 2 + k];
    int pos = atomicAdd(&fill[e], 1);
    int s = offs[e] + pos;
    slot_token[s] = t;
    slot_w[s] = top_w[t * 2 + k];
  }
}

// ---------------------------------------------------------------------------
extern "C" void kernel_launch(void* const* d_in, const int* in_sizes, int n_in,
                              void* d_out, int out_size, void* d_ws, size_t ws_size,
                              hipStream_t stream)
{
  (void)in_sizes; (void)n_in; (void)out_size; (void)ws_size;
  const float* x      = (const float*)d_in[0];
  const float* tsq    = (const float*)d_in[1];
  const float* tsk    = (const float*)d_in[2];
  const float* anw    = (const float*)d_in[3];
  const float* fnw    = (const float*)d_in[4];
  const float* w_kv_c = (const float*)d_in[5];
  const float* w_kc   = (const float*)d_in[6];
  const float* w_vc   = (const float*)d_in[7];
  const float* w_qr   = (const float*)d_in[8];
  const float* w_kr   = (const float*)d_in[9];
  const float* w_o    = (const float*)d_in[10];
  const float* gw     = (const float*)d_in[11];
  const float* ebias  = (const float*)d_in[12];
  const float* w1     = (const float*)d_in[13];
  const float* w2     = (const float*)d_in[14];
  float* outF = (float*)d_out;

  uint8_t* p = (uint8_t*)d_ws;
  auto alloc = [&](size_t bytes) -> uint8_t* {
    uint8_t* r = p; p += (bytes + 255) & ~(size_t)255; return r;
  };
  unsigned short* h_bf    = (unsigned short*)alloc((size_t)T_TOK * DMODEL * 2);
  unsigned short* ckv_bf  = (unsigned short*)alloc((size_t)T_TOK * DC * 2);
  unsigned short* kc_bf   = (unsigned short*)alloc((size_t)T_TOK * DMODEL * 2);
  unsigned short* vc_bf   = (unsigned short*)alloc((size_t)T_TOK * DMODEL * 2);
  unsigned short* qr_bf   = (unsigned short*)alloc((size_t)T_TOK * DMODEL * 2);
  unsigned short* kr_bf   = (unsigned short*)alloc((size_t)T_TOK * DMODEL * 2);
  unsigned short* q_bf    = (unsigned short*)alloc((size_t)T_TOK * DMODEL * 2);
  unsigned short* ksum_bf = (unsigned short*)alloc((size_t)T_TOK * DMODEL * 2);
  unsigned short* vt_bf   = (unsigned short*)alloc((size_t)T_TOK * DMODEL * 2);
  unsigned short* ao_bf   = (unsigned short*)alloc((size_t)T_TOK * DMODEL * 2);
  float*          x1_f    = (float*)alloc((size_t)T_TOK * DMODEL * 4);
  unsigned short* hn_bf   = (unsigned short*)alloc((size_t)T_TOK * DMODEL * 2);
  unsigned short* hmid_bf = (unsigned short*)alloc((size_t)MAXPAIRS * DFF2 * 2);
  unsigned short* wb_kvc = (unsigned short*)alloc((size_t)DC * DMODEL * 2);
  unsigned short* wb_kc  = (unsigned short*)alloc((size_t)DMODEL * DC * 2);
  unsigned short* wb_vc  = (unsigned short*)alloc((size_t)DMODEL * DC * 2);
  unsigned short* wb_qr  = (unsigned short*)alloc((size_t)DMODEL * DMODEL * 2);
  unsigned short* wb_kr  = (unsigned short*)alloc((size_t)DMODEL * DMODEL * 2);
  unsigned short* wb_o   = (unsigned short*)alloc((size_t)DMODEL * DMODEL * 2);
  unsigned short* wb_1   = (unsigned short*)alloc((size_t)NEXP * DFF2 * DMODEL * 2);
  unsigned short* wb_2   = (unsigned short*)alloc((size_t)NEXP * DMODEL * DFF2 * 2);
  int*   top_i      = (int*)alloc((size_t)T_TOK * 2 * 4);
  float* top_w      = (float*)alloc((size_t)T_TOK * 2 * 4);
  int*   slot_token = (int*)alloc((size_t)MAXPAIRS * 4);
  float* slot_w     = (float*)alloc((size_t)MAXPAIRS * 4);
  int*   counts     = (int*)alloc(NEXP * 4);
  int*   offs       = (int*)alloc(NEXP * 4);
  int*   fill       = (int*)alloc(NEXP * 4);

  hipMemsetAsync(counts, 0, NEXP * 4, stream);
  hipMemsetAsync(fill,   0, NEXP * 4, stream);
  hipMemsetAsync(outF + (size_t)T_TOK * DMODEL, 0, NEXP * 4, stream);

  // 0) one-time weight conversion fp32 -> bf16
  f32_to_bf16_kernel<<<512, 256, 0, stream>>>(w_kv_c, wb_kvc, DC * DMODEL);
  f32_to_bf16_kernel<<<512, 256, 0, stream>>>(w_kc,   wb_kc,  DMODEL * DC);
  f32_to_bf16_kernel<<<512, 256, 0, stream>>>(w_vc,   wb_vc,  DMODEL * DC);
  f32_to_bf16_kernel<<<1024, 256, 0, stream>>>(w_qr,  wb_qr,  DMODEL * DMODEL);
  f32_to_bf16_kernel<<<1024, 256, 0, stream>>>(w_kr,  wb_kr,  DMODEL * DMODEL);
  f32_to_bf16_kernel<<<1024, 256, 0, stream>>>(w_o,   wb_o,   DMODEL * DMODEL);
  f32_to_bf16_kernel<<<8192, 256, 0, stream>>>(w1, wb_1, NEXP * DFF2 * DMODEL);
  f32_to_bf16_kernel<<<8192, 256, 0, stream>>>(w2, wb_2, NEXP * DMODEL * DFF2);

  // 1) attention-input RMSNorm
  rmsnorm_kernel<<<T_TOK, 256, 0, stream>>>(x, anw, h_bf, nullptr);

  // 2) projections (C = A * W^T, bf16 WMMA, 128x64 tiles)
  gemm_bf16<0><<<dim3(DC / 64, T_TOK / 128, 1), 128, 0, stream>>>(
      h_bf, wb_kvc, T_TOK, DC, DMODEL, ckv_bf, nullptr, nullptr, nullptr, nullptr, nullptr, nullptr);
  gemm_bf16<0><<<dim3(DMODEL / 64, T_TOK / 128, 1), 128, 0, stream>>>(
      ckv_bf, wb_kc, T_TOK, DMODEL, DC, kc_bf, nullptr, nullptr, nullptr, nullptr, nullptr, nullptr);
  gemm_bf16<0><<<dim3(DMODEL / 64, T_TOK / 128, 1), 128, 0, stream>>>(
      ckv_bf, wb_vc, T_TOK, DMODEL, DC, vc_bf, nullptr, nullptr, nullptr, nullptr, nullptr, nullptr);
  gemm_bf16<0><<<dim3(DMODEL / 64, T_TOK / 128, 1), 128, 0, stream>>>(
      h_bf, wb_qr, T_TOK, DMODEL, DMODEL, qr_bf, nullptr, nullptr, nullptr, nullptr, nullptr, nullptr);
  gemm_bf16<0><<<dim3(DMODEL / 64, T_TOK / 128, 1), 128, 0, stream>>>(
      h_bf, wb_kr, T_TOK, DMODEL, DMODEL, kr_bf, nullptr, nullptr, nullptr, nullptr, nullptr, nullptr);

  // 3) RoPE fuse
  rope_kernel<<<(T_TOK * NHEADS * 32) / 256, 256, 0, stream>>>(
      qr_bf, kr_bf, kc_bf, tsq, tsk, q_bf, ksum_bf);

  // 4) V transpose
  transpose_v_kernel<<<dim3(L_SEQ / 16, HDIM / 16, BATCH * NHEADS), dim3(16, 16), 0, stream>>>(
      vc_bf, vt_bf);

  // 5) flash attention
  attn_kernel<<<dim3(L_SEQ / 64, BATCH * NHEADS), 128, 0, stream>>>(
      q_bf, ksum_bf, vt_bf, ao_bf);

  // 6) output projection + residual -> x1 (fp32)
  gemm_bf16<1><<<dim3(DMODEL / 64, T_TOK / 128, 1), 128, 0, stream>>>(
      ao_bf, wb_o, T_TOK, DMODEL, DMODEL, nullptr, x1_f, x, nullptr, nullptr, nullptr, nullptr);

  // 7) FFN RMSNorm (+copy x1 into d_out as residual base)
  rmsnorm_kernel<<<T_TOK, 256, 0, stream>>>(x1_f, fnw, hn_bf, outF);

  // 8) routing
  router_kernel<<<T_TOK, 32, 0, stream>>>(hn_bf, gw, ebias, top_i, top_w, counts,
                                          outF + (size_t)T_TOK * DMODEL);
  scan_kernel<<<1, 32, 0, stream>>>(counts, offs);
  scatter_kernel<<<T_TOK / 256, 256, 0, stream>>>(top_i, top_w, offs, fill, slot_token, slot_w);

  // 9) MoE expert GEMMs (routed top-2, identical to reference's masked-dense)
  gemm_bf16<2><<<dim3(DFF2 / 64, T_TOK / 128, NEXP), 128, 0, stream>>>(
      hn_bf, wb_1, 0, DFF2, DMODEL, hmid_bf, nullptr, nullptr, slot_token, slot_w, counts, offs);
  gemm_bf16<3><<<dim3(DMODEL / 64, T_TOK / 128, NEXP), 128, 0, stream>>>(
      hmid_bf, wb_2, 0, DMODEL, DFF2, nullptr, outF, nullptr, slot_token, slot_w, counts, offs);
}
